// RW_GNN_85993835200811
// MI455X (gfx1250) — matrix operand order repaired and checked by reading, not compiled, not run on hardware
//
#include <hip/hip_runtime.h>

typedef _Float16 v16h __attribute__((ext_vector_type(16)));
typedef float    v8f  __attribute__((ext_vector_type(8)));

#define N_GRAPHS 128
#define N_SUB    32
#define FEAT     96    // N_SUB * (MAX_STEP-1)
#define HID      128

// ---------------------------------------------------------------------------
// u_out[row] = sum_col adj[row][col] * u_in[col]   (u_in == nullptr -> ones)
// one wave32 per row, float4 (b128) coalesced loads, shuffle reduction
// ---------------------------------------------------------------------------
__global__ void matvec_kernel(const float* __restrict__ adj,
                              const float* __restrict__ uin,
                              float* __restrict__ uout, int n) {
    int wid  = (blockIdx.x * blockDim.x + threadIdx.x) >> 5;
    int lane = threadIdx.x & 31;
    if (wid >= n) return;
    const float* row = adj + (size_t)wid * n;
    float acc = 0.f;
    if (uin) {
        for (int c = lane * 4; c < n; c += 128) {
            float4 a = *(const float4*)(row + c);
            float4 u = *(const float4*)(uin + c);
            acc += a.x * u.x + a.y * u.y + a.z * u.z + a.w * u.w;
        }
    } else {
        for (int c = lane * 4; c < n; c += 128) {
            float4 a = *(const float4*)(row + c);
            acc += a.x + a.y + a.z + a.w;
        }
    }
    for (int off = 16; off; off >>= 1) acc += __shfl_xor(acc, off, 32);
    if (lane == 0) uout[wid] = acc;
}

// ---------------------------------------------------------------------------
// S_i[s] = 1^T (A_sub^i) 1  for i = 1..3 ; A_sub from relu(theta), symmetric
// ---------------------------------------------------------------------------
__global__ void sub_kernel(const float* __restrict__ theta,
                           float* __restrict__ Ssum) {
    int s = threadIdx.x;
    if (s >= N_SUB) return;
    float A[100];
    #pragma unroll
    for (int i = 0; i < 100; i++) A[i] = 0.f;
    const float* th = theta + s * 45;
    int idx = 0;
    for (int i = 0; i < 10; i++)
        for (int j = i + 1; j < 10; j++) {
            float v = th[idx++];
            v = v > 0.f ? v : 0.f;
            A[i * 10 + j] = v;
            A[j * 10 + i] = v;
        }
    float w[10];
    #pragma unroll
    for (int i = 0; i < 10; i++) w[i] = 1.f;
    for (int step = 0; step < 3; step++) {
        float nw[10], tot = 0.f;
        for (int i = 0; i < 10; i++) {
            float a = 0.f;
            for (int j = 0; j < 10; j++) a += A[i * 10 + j] * w[j];
            nw[i] = a;
            tot += a;
        }
        #pragma unroll
        for (int i = 0; i < 10; i++) w[i] = nw[i];
        Ssum[step * N_SUB + s] = tot;
    }
}

// ---------------------------------------------------------------------------
// Deterministic segment sums: gsum[i][g] = sum_{m: gi[m]==g} u_i[m], counts[g]
// one block per graph, fixed shared-memory reduction tree (graph-replay safe)
// ---------------------------------------------------------------------------
__global__ void segsum_kernel(const float* __restrict__ u,
                              const int* __restrict__ gi,
                              float* __restrict__ gsum,
                              float* __restrict__ counts, int n) {
    int g = blockIdx.x, t = threadIdx.x;   // 128 blocks x 128 threads
    float s0 = 0.f, s1 = 0.f, s2 = 0.f, c = 0.f;
    for (int i = t; i < n; i += 128) {
        if (gi[i] == g) {
            s0 += u[i];
            s1 += u[n + i];
            s2 += u[2 * n + i];
            c  += 1.f;
        }
    }
    __shared__ float sh[4][128];
    sh[0][t] = s0; sh[1][t] = s1; sh[2][t] = s2; sh[3][t] = c;
    __syncthreads();
    for (int off = 64; off; off >>= 1) {
        if (t < off)
            for (int r = 0; r < 4; r++) sh[r][t] += sh[r][t + off];
        __syncthreads();
    }
    if (t == 0) {
        gsum[g]                = sh[0][0];
        gsum[N_GRAPHS + g]     = sh[1][0];
        gsum[2 * N_GRAPHS + g] = sh[2][0];
        counts[g]              = sh[3][0];
    }
}

// x[g, f] = Ssum[f] * gsum[step][g] / counts[g],  step = f/32
__global__ void feat_kernel(const float* __restrict__ Ssum,
                            const float* __restrict__ gsum,
                            const float* __restrict__ counts,
                            float* __restrict__ x) {
    int idx = blockIdx.x * blockDim.x + threadIdx.x;
    if (idx >= N_GRAPHS * FEAT) return;
    int g = idx / FEAT, f = idx % FEAT;
    int step = f >> 5;
    x[idx] = Ssum[f] * gsum[step * N_GRAPHS + g] / counts[g];
}

// ---------------------------------------------------------------------------
// BatchNorm over the batch (128 graphs) per feature, then split normalized
// activation into hi/lo f16 halves for the split-precision WMMA GEMM.
// ---------------------------------------------------------------------------
__global__ void bn_kernel(const float* __restrict__ x,
                          const float* __restrict__ gamma,
                          const float* __restrict__ beta,
                          _Float16* __restrict__ xh,
                          _Float16* __restrict__ xl) {
    int f = blockIdx.x;     // 0..95
    int g = threadIdx.x;    // 0..127
    float v = x[g * FEAT + f];
    __shared__ float sh[128];
    sh[g] = v;
    __syncthreads();
    for (int off = 64; off; off >>= 1) {
        if (g < off) sh[g] += sh[g + off];
        __syncthreads();
    }
    float mean = sh[0] * (1.f / N_GRAPHS);
    __syncthreads();
    float d = v - mean;
    sh[g] = d * d;
    __syncthreads();
    for (int off = 64; off; off >>= 1) {
        if (g < off) sh[g] += sh[g + off];
        __syncthreads();
    }
    float var  = sh[0] * (1.f / N_GRAPHS);
    float rstd = rsqrtf(var + 1e-5f);
    float xa   = gamma[f] * d * rstd + beta[f];
    _Float16 hi = (_Float16)xa;
    xh[g * FEAT + f] = hi;
    xl[g * FEAT + f] = (_Float16)(xa - (float)hi);
}

// fp32 -> hi/lo f16 split of weights
__global__ void split_kernel(const float* __restrict__ w,
                             _Float16* __restrict__ wh,
                             _Float16* __restrict__ wl, int n) {
    int i = blockIdx.x * blockDim.x + threadIdx.x;
    if (i >= n) return;
    float v = w[i];
    _Float16 hi = (_Float16)v;
    wh[i] = hi;
    wl[i] = (_Float16)(v - (float)hi);
}

// ---------------------------------------------------------------------------
// h = relu(xhat @ w1 + b1), M=N=128, K=96 via v_wmma_f32_16x16x32_f16.
// Split-precision: acc += Ah*Bh + Ah*Bl + Al*Bh  (~fp32 accuracy).
// One wave per 16x16 tile, 64 tiles = 8 blocks x 8 waves; EXEC all ones.
// ---------------------------------------------------------------------------
__global__ void gemm1_kernel(const _Float16* __restrict__ xh,
                             const _Float16* __restrict__ xl,
                             const _Float16* __restrict__ wh,
                             const _Float16* __restrict__ wl,
                             const float* __restrict__ b1,
                             float* __restrict__ h) {
    int wid  = (blockIdx.x * blockDim.x + threadIdx.x) >> 5;  // 0..63
    int lane = threadIdx.x & 31;
    int tm = wid >> 3, tn = wid & 7;
    int m0 = tm * 16, n0 = tn * 16;
    int half = lane >> 4;     // 0: lanes 0-15, 1: lanes 16-31
    int l15  = lane & 15;

    v8f acc = {};
    for (int kt = 0; kt < 3; kt++) {
        int k0 = kt * 32;
        v16h ah, al, bh, bl;
        // A tile: M = m0+l15 ; ISA layout: halves 0-7 -> K = k0 + half*8 + t,
        //                                  halves 8-15 -> K = k0 + 16 + half*8 + t
        const _Float16* rh = xh + (m0 + l15) * FEAT;
        const _Float16* rl = xl + (m0 + l15) * FEAT;
        int ka = k0 + half * 8;
        #pragma unroll
        for (int t = 0; t < 8; t++) {
            ah[t]     = rh[ka + t];
            al[t]     = rl[ka + t];
            ah[8 + t] = rh[ka + 16 + t];
            al[8 + t] = rl[ka + 16 + t];
        }
        // B tile: N = n0+l15 ; same K layout as A: 0-7 -> k0+half*8+t, 8-15 -> k0+16+half*8+t
        int kb = k0 + half * 8;
        #pragma unroll
        for (int t = 0; t < 16; t++) {
            int kk = kb + t + (t >= 8 ? 8 : 0);
            bh[t] = wh[kk * HID + n0 + l15];
            bl[t] = wl[kk * HID + n0 + l15];
        }
        acc = __builtin_amdgcn_wmma_f32_16x16x32_f16(false, ah, false, bh,
                                                     (short)0, acc, false, false);
        acc = __builtin_amdgcn_wmma_f32_16x16x32_f16(false, ah, false, bl,
                                                     (short)0, acc, false, false);
        acc = __builtin_amdgcn_wmma_f32_16x16x32_f16(false, al, false, bh,
                                                     (short)0, acc, false, false);
    }
    // C/D layout: VGPR r holds M = m0 + r + half*8, N = n0 + l15
    int N = n0 + l15;
    float bias = b1[N];
    #pragma unroll
    for (int r = 0; r < 8; r++) {
        int M = m0 + r + half * 8;
        float v = acc[r] + bias;
        h[M * HID + N] = v > 0.f ? v : 0.f;
    }
}

// out = h @ w2 + b2 : [128,128]@[128,2] -> one thread per output element
__global__ void mlp2_kernel(const float* __restrict__ h,
                            const float* __restrict__ w2,
                            const float* __restrict__ b2,
                            float* __restrict__ out) {
    int tid = threadIdx.x;          // 256 = 128 graphs x 2 outputs
    int g = tid >> 1, o = tid & 1;
    float acc = b2[o];
    for (int j = 0; j < HID; j++) acc += h[g * HID + j] * w2[j * 2 + o];
    out[g * 2 + o] = acc;
}

extern "C" void kernel_launch(void* const* d_in, const int* in_sizes, int n_in,
                              void* d_out, int out_size, void* d_ws, size_t ws_size,
                              hipStream_t stream) {
    const float* adj   = (const float*)d_in[0];
    const int*   gi    = (const int*)  d_in[1];
    const float* theta = (const float*)d_in[2];
    const float* gamma = (const float*)d_in[3];
    const float* beta  = (const float*)d_in[4];
    const float* w1    = (const float*)d_in[5];
    const float* b1    = (const float*)d_in[6];
    const float* w2    = (const float*)d_in[7];
    const float* b2    = (const float*)d_in[8];
    float* out = (float*)d_out;
    const int n = in_sizes[1];      // n_nodes = 2048

    // workspace carve-up (all fully written before read; no init needed)
    char*  ws     = (char*)d_ws;
    float* u      = (float*)ws;                          // 3*n
    float* gsum   = (float*)(ws + (size_t)3 * n * 4);    // 3*128
    float* counts = gsum + 3 * N_GRAPHS;                 // 128
    float* Ssum   = counts + N_GRAPHS;                   // 96
    float* x      = Ssum + FEAT;                         // 128*96
    float* h      = x + N_GRAPHS * FEAT;                 // 128*128
    _Float16* xh  = (_Float16*)(h + N_GRAPHS * HID);     // 128*96
    _Float16* xl  = xh + N_GRAPHS * FEAT;
    _Float16* wh  = xl + N_GRAPHS * FEAT;                // 96*128
    _Float16* wl  = wh + FEAT * HID;

    int mv_blocks = (n + 7) / 8;    // one wave32 per row, 8 waves/block

    sub_kernel   <<<1, 32, 0, stream>>>(theta, Ssum);
    matvec_kernel<<<mv_blocks, 256, 0, stream>>>(adj, (const float*)nullptr, u, n);
    matvec_kernel<<<mv_blocks, 256, 0, stream>>>(adj, u,         u + n,     n);
    matvec_kernel<<<mv_blocks, 256, 0, stream>>>(adj, u + n,     u + 2 * n, n);
    segsum_kernel<<<N_GRAPHS, 128, 0, stream>>>(u, gi, gsum, counts, n);
    feat_kernel  <<<(N_GRAPHS * FEAT + 255) / 256, 256, 0, stream>>>(Ssum, gsum, counts, x);
    bn_kernel    <<<FEAT, N_GRAPHS, 0, stream>>>(x, gamma, beta, xh, xl);
    split_kernel <<<(FEAT * HID + 255) / 256, 256, 0, stream>>>(w1, wh, wl, FEAT * HID);
    gemm1_kernel <<<8, 256, 0, stream>>>(xh, xl, wh, wl, b1, h);
    mlp2_kernel  <<<1, 256, 0, stream>>>(h, w2, b2, out);
}
